// Attention2d_10797547782848
// MI455X (gfx1250) — matrix operand order repaired and checked by reading, not compiled, not run on hardware
//
#include <hip/hip_runtime.h>

// ---------------------------------------------------------------------------
// CDNA5 (gfx1250) fused 2D relative-position attention
//   proj(Q,K,V) -> flash attention (softmax over k) -> proj(U)
// Matrix math: v_wmma_f32_16x16x32_bf16 (hc=32 == WMMA K-dim).
// Activation tiles staged into LDS via the Tensor Data Mover (TDM).
// ---------------------------------------------------------------------------

typedef __bf16 v16bf __attribute__((ext_vector_type(16)));
typedef __bf16 v8bf  __attribute__((ext_vector_type(8)));
typedef float  v8f   __attribute__((ext_vector_type(8)));
typedef unsigned int u32x4 __attribute__((ext_vector_type(4)));
typedef int          i32x4 __attribute__((ext_vector_type(4)));
typedef int          i32x8 __attribute__((ext_vector_type(8)));

#define B_BATCH 16
#define E_DIM   128
#define H_HEADS 4
#define HC      32
#define NYD     32
#define NXD     32
#define N_TOK   (NYD * NXD)           // 1024
#define PE_LEN  ((2*NYD-1)*(2*NXD-1)) // 3969

__device__ __forceinline__ v8f wmma_bf16(v16bf a, v16bf b, v8f c) {
  return __builtin_amdgcn_wmma_f32_16x16x32_bf16(false, a, false, b,
                                                 (short)0, c, false, false);
}

// ---------------------------------------------------------------------------
// Projection GEMM: Y[o,n] = sum_c W[o,c] * X[c,n] + bias[o]   (per batch)
// Block = 256 threads (8 waves). Workgroup tile: 128 o  x  64 n.
// W (128x128 -> bf16) staged by VALU; X tile (128c x 64n, native dtype)
// staged by one TDM tensor_load_to_lds (TENSORcnt), then workgroup barrier.
// ---------------------------------------------------------------------------
template <typename Tin, bool OUT_BF16, bool TRANS_OUT>
__global__ __launch_bounds__(256)
void proj_kernel(const Tin* __restrict__ X, const float* __restrict__ W,
                 const float* __restrict__ bias, void* __restrict__ Yv)
{
  __shared__ __bf16 Wl[E_DIM * E_DIM];  // [o][c], 32 KB
  __shared__ Tin    Xl[E_DIM * 64];     // [c][n] tile, native dtype

  const int tid  = threadIdx.x;
  const int nblk = blockIdx.x;          // 0..15 (tile of 64 tokens)
  const int b    = blockIdx.y;          // batch
  const int n0   = nblk * 64;

  // stage weights (f32 -> bf16)
  for (int i = tid; i < E_DIM * E_DIM; i += 256)
    Wl[i] = (__bf16)W[i];

#if __has_builtin(__builtin_amdgcn_tensor_load_to_lds)
  // --- TDM: DMA the 2D activation tile [128 c][64 n] into LDS ------------
  if (tid < 32) {                        // one wave issues the descriptor
    const unsigned long long ga =
        (unsigned long long)(const void*)(X + (size_t)b * E_DIM * N_TOK + n0);
    const unsigned ldsoff = (unsigned)(size_t)(void*)Xl;  // LDS byte offset
    const unsigned dsz    = (sizeof(Tin) == 4) ? 2u : 1u; // 4B / 2B elements
    const unsigned td0 = N_TOK, td1 = E_DIM;              // tensor dims (elems)
    const unsigned tl0 = 64,    tl1 = E_DIM;              // tile dims   (elems)
    const unsigned s0  = N_TOK;                           // dim0 stride (elems)

    u32x4 g0;
    g0[0] = 1u;                                           // count=1
    g0[1] = ldsoff;                                       // lds_addr
    g0[2] = (unsigned)(ga & 0xFFFFFFFFu);                 // global_addr lo
    g0[3] = (unsigned)((ga >> 32) & 0x1FFFFFFu) | 0x80000000u;  // hi | type=2

    i32x8 g1;
    g1[0] = (int)(dsz << 16);                             // data_size
    g1[1] = (int)((td0 & 0xFFFFu) << 16);                 // tensor_dim0 lo
    g1[2] = (int)(((td0 >> 16) & 0xFFFFu) | ((td1 & 0xFFFFu) << 16));
    g1[3] = (int)(((td1 >> 16) & 0xFFFFu) | ((tl0 & 0xFFFFu) << 16));
    g1[4] = (int)(tl1 & 0xFFFFu);                         // tile_dim1, dim2=0
    g1[5] = (int)s0;                                      // dim0_stride lo
    g1[6] = 0;                                            // stride hi / dim1_stride
    g1[7] = 0;

    i32x4 z4 = {};
#if __clang_major__ >= 23
    i32x8 z8 = {};
    __builtin_amdgcn_tensor_load_to_lds(g0, g1, z4, z4, z8, 0);
#else
    __builtin_amdgcn_tensor_load_to_lds(g0, g1, z4, z4, 0);
#endif
    __builtin_amdgcn_s_wait_tensorcnt(0);
  }
#else
  // Fallback: cooperative staging
  for (int i = tid; i < E_DIM * 64; i += 256) {
    int c = i >> 6, n = i & 63;
    Xl[i] = X[(b * E_DIM + c) * N_TOK + n0 + n];
  }
#endif
  __syncthreads();

  const int wave  = tid >> 5;
  const int lane  = tid & 31;
  const int ohalf = wave >> 2;               // 0..1
  const int nsub  = wave & 3;                // 0..3
  const int ncol  = nsub * 16 + (lane & 15); // column within 64-tile
  const int coff  = (lane < 16) ? 0 : 8;     // A-operand K-half select

  v8f acc[4] = {v8f{}, v8f{}, v8f{}, v8f{}};

  for (int cc = 0; cc < E_DIM; cc += 32) {
    // B operand: 32c x 16n; lane j holds col ncol, K = (j/16)*16 + i
    v16bf bv;
    {
      const int cb = cc + (lane >> 4) * 16;
#pragma unroll
      for (int i = 0; i < 16; ++i)
        bv[i] = (__bf16)(float)Xl[(cb + i) * 64 + ncol];
    }
#pragma unroll
    for (int ot = 0; ot < 4; ++ot) {
      const int orow = ohalf * 64 + ot * 16 + (lane & 15);
      v16bf av;  // A: 16o x 32c, halves {2*coff..2*coff+15}
#pragma unroll
      for (int i = 0; i < 8; ++i) {
        av[i]     = Wl[orow * E_DIM + cc + 2 * coff + i];
        av[i + 8] = Wl[orow * E_DIM + cc + 2 * coff + 8 + i];
      }
      acc[ot] = wmma_bf16(av, bv, acc[ot]);
    }
  }

#pragma unroll
  for (int ot = 0; ot < 4; ++ot) {
#pragma unroll
    for (int r = 0; r < 8; ++r) {
      const int o = ohalf * 64 + ot * 16 + r + ((lane < 16) ? 0 : 8);
      const int n = n0 + ncol;
      const float y = acc[ot][r] + bias[o];
      if (OUT_BF16) {
        __bf16* Y = (__bf16*)Yv;
        if (TRANS_OUT) Y[((size_t)b * N_TOK + n) * E_DIM + o] = (__bf16)y;
        else           Y[((size_t)b * E_DIM + o) * N_TOK + n] = (__bf16)y;
      } else {
        ((float*)Yv)[((size_t)b * E_DIM + o) * N_TOK + n] = y;
      }
    }
  }
}

// ---------------------------------------------------------------------------
// Fused attention, flash-style along k (softmax axis).
// Grid: (N/64, H, B). Block = 128 (4 waves); wave w owns 16 q columns.
// Per 32-k chunk: 2 score WMMAs, rel-bias gather from LDS, online softmax
// (cross-half via 2 reduce + 4 packed-bf16 shuffles), 2 output WMMAs.
// ---------------------------------------------------------------------------
__global__ __launch_bounds__(128)
void attn_kernel(const __bf16* __restrict__ Qb,   // [b][e][n]
                 const __bf16* __restrict__ Kt,   // [b][n][e] (transposed)
                 const __bf16* __restrict__ Vb,   // [b][e][n]
                 const float*  __restrict__ pos_enc,
                 __bf16* __restrict__ Ob)         // [b][e][n]
{
  __shared__ float pe_s[PE_LEN];

  const int tid = threadIdx.x;
  const int h   = blockIdx.y;
  const int b   = blockIdx.z;
  for (int i = tid; i < PE_LEN; i += 128)
    pe_s[i] = pos_enc[h * PE_LEN + i];
  __syncthreads();

  const int wave  = tid >> 5;
  const int lane  = tid & 31;
  const int q0    = blockIdx.x * 64 + wave * 16;
  const int qcol  = q0 + (lane & 15);
  const int yq    = qcol >> 5, xq = qcol & 31;
  const int ebase = h * HC;
  const int coff  = (lane < 16) ? 0 : 8;
  const bool loHalf = (lane < 16);
  const float scale = 0.17677669529663687f;  // 1/sqrt(32)

  // Persistent Q B-operand: 32c x 16q
  v16bf qv;
  {
    const int cb = ebase + (lane >> 4) * 16;
#pragma unroll
    for (int i = 0; i < 16; ++i)
      qv[i] = Qb[((size_t)b * E_DIM + cb + i) * N_TOK + qcol];
  }

  v8f o0 = v8f{}, o1 = v8f{};
  float m = -__builtin_inff(), l = 0.f;

  const __bf16* kp0 =
      &Kt[((size_t)b * N_TOK + (lane & 15)) * E_DIM + ebase];
  const __bf16* vp0 =
      &Vb[((size_t)b * E_DIM + ebase + (lane & 15)) * N_TOK];

  for (int k0 = 0; k0 < N_TOK; k0 += 32) {
    // prefetch next chunk's K / V lines into cache
    if (k0 + 32 < N_TOK) {
      __builtin_prefetch(kp0 + (size_t)(k0 + 32) * E_DIM);
      __builtin_prefetch(vp0 + k0 + 32);
    }

    // --- scores: S[k,q] = sum_c K[c,k] Q[c,q] ---
    v16bf a0, a1;  // K^T tiles: rows = tokens k0..+15 / k0+16..+31, K-dim = c
    {
      const __bf16* k0p = kp0 + (size_t)k0 * E_DIM;
      const __bf16* k1p = k0p + 16 * E_DIM;
      v8bf lo0 = *(const v8bf*)(k0p + 2 * coff);
      v8bf hi0 = *(const v8bf*)(k0p + 2 * coff + 8);
      v8bf lo1 = *(const v8bf*)(k1p + 2 * coff);
      v8bf hi1 = *(const v8bf*)(k1p + 2 * coff + 8);
#pragma unroll
      for (int i = 0; i < 8; ++i) {
        a0[i] = lo0[i]; a0[i + 8] = hi0[i];
        a1[i] = lo1[i]; a1[i + 8] = hi1[i];
      }
    }
    v8f s0 = wmma_bf16(a0, qv, v8f{});
    v8f s1 = wmma_bf16(a1, qv, v8f{});

    // --- scale + relative position bias + chunk max ---
    float e0[8], e1[8];
    float mx = -__builtin_inff();
#pragma unroll
    for (int r = 0; r < 8; ++r) {
      const int kr0 = k0 + r + (loHalf ? 0 : 8);
      const int kr1 = kr0 + 16;
      const float r0 =
          pe_s[((kr0 >> 5) - yq + NYD - 1) * (2 * NXD - 1) + (kr0 & 31) - xq + NXD - 1];
      const float r1 =
          pe_s[((kr1 >> 5) - yq + NYD - 1) * (2 * NXD - 1) + (kr1 & 31) - xq + NXD - 1];
      e0[r] = s0[r] * scale + r0;
      e1[r] = s1[r] * scale + r1;
      mx = fmaxf(mx, fmaxf(e0[r], e1[r]));
    }
    mx = fmaxf(mx, __shfl_xor(mx, 16, 32));  // combine k-halves of this column

    // --- online softmax update ---
    const float m_new   = fmaxf(m, mx);
    const float rescale = __expf(m - m_new);
    float ssum = 0.f;
#pragma unroll
    for (int r = 0; r < 8; ++r) {
      e0[r] = __expf(e0[r] - m_new);
      e1[r] = __expf(e1[r] - m_new);
      ssum += e0[r] + e1[r];
    }
    ssum += __shfl_xor(ssum, 16, 32);
    l = l * rescale + ssum;
    m = m_new;
#pragma unroll
    for (int r = 0; r < 8; ++r) { o0[r] *= rescale; o1[r] *= rescale; }

    // --- pack P (32k x 16q) as WMMA B operand -------------------------------
    // Each lane sends exactly what its half-partner needs (lanes<16 send e1,
    // lanes>=16 send e0) as packed-bf16 pairs: 4 shuffles instead of 16.
    unsigned rcv[4];
#pragma unroll
    for (int j = 0; j < 4; ++j) {
      const float slo = loHalf ? e1[2 * j]     : e0[2 * j];
      const float shi = loHalf ? e1[2 * j + 1] : e0[2 * j + 1];
      const unsigned snd =
          (unsigned)__builtin_bit_cast(unsigned short, (__bf16)slo) |
          ((unsigned)__builtin_bit_cast(unsigned short, (__bf16)shi) << 16);
      rcv[j] = __shfl_xor(snd, 16, 32);
    }
    v16bf pv;
#pragma unroll
    for (int r = 0; r < 8; ++r) {
      const __bf16 t = __builtin_bit_cast(
          __bf16, (unsigned short)((r & 1) ? (rcv[r >> 1] >> 16)
                                           : (rcv[r >> 1] & 0xFFFFu)));
      pv[r]     = loHalf ? (__bf16)e0[r] : t;
      pv[r + 8] = loHalf ? t             : (__bf16)e1[r];
    }

    // --- O += V * P ---
    v16bf va0, va1;  // V tiles: rows = channels 0..15 / 16..31, K-dim = k
    {
      const __bf16* v0p = vp0 + k0;
      const __bf16* v1p = v0p + 16 * N_TOK;
      v8bf lo0 = *(const v8bf*)(v0p + 2 * coff);
      v8bf hi0 = *(const v8bf*)(v0p + 2 * coff + 8);
      v8bf lo1 = *(const v8bf*)(v1p + 2 * coff);
      v8bf hi1 = *(const v8bf*)(v1p + 2 * coff + 8);
#pragma unroll
      for (int i = 0; i < 8; ++i) {
        va0[i] = lo0[i]; va0[i + 8] = hi0[i];
        va1[i] = lo1[i]; va1[i + 8] = hi1[i];
      }
    }
    o0 = wmma_bf16(va0, pv, o0);
    o1 = wmma_bf16(va1, pv, o1);
  }

  // --- normalize + store (channel-major bf16) ---
  const float inv = 1.0f / l;
#pragma unroll
  for (int r = 0; r < 8; ++r) {
    const int c = r + (loHalf ? 0 : 8);
    Ob[((size_t)b * E_DIM + ebase + c) * N_TOK + qcol]      = (__bf16)(o0[r] * inv);
    Ob[((size_t)b * E_DIM + ebase + c + 16) * N_TOK + qcol] = (__bf16)(o1[r] * inv);
  }
}

// ---------------------------------------------------------------------------
extern "C" void kernel_launch(void* const* d_in, const int* in_sizes, int n_in,
                              void* d_out, int out_size, void* d_ws, size_t ws_size,
                              hipStream_t stream) {
  (void)in_sizes; (void)n_in; (void)out_size; (void)ws_size;
  const float* x  = (const float*)d_in[0];
  const float* Wk = (const float*)d_in[1];
  const float* bk = (const float*)d_in[2];
  const float* Wq = (const float*)d_in[3];
  const float* bq = (const float*)d_in[4];
  const float* Wv = (const float*)d_in[5];
  const float* bv = (const float*)d_in[6];
  const float* Wu = (const float*)d_in[7];
  const float* bu = (const float*)d_in[8];
  const float* pe = (const float*)d_in[9];
  float* out = (float*)d_out;

  char* ws = (char*)d_ws;
  const size_t buf = (size_t)B_BATCH * E_DIM * N_TOK * sizeof(__bf16);  // 4 MB
  __bf16* Qb = (__bf16*)(ws);
  __bf16* Kt = (__bf16*)(ws + buf);
  __bf16* Vb = (__bf16*)(ws + 2 * buf);
  __bf16* Ob = (__bf16*)(ws + 3 * buf);

  dim3 pgrid(N_TOK / 64, B_BATCH), pblk(256);
  proj_kernel<float,  true,  false><<<pgrid, pblk, 0, stream>>>(x,  Wq, bq, (void*)Qb);
  proj_kernel<float,  true,  true ><<<pgrid, pblk, 0, stream>>>(x,  Wk, bk, (void*)Kt);
  proj_kernel<float,  true,  false><<<pgrid, pblk, 0, stream>>>(x,  Wv, bv, (void*)Vb);

  dim3 agrid(N_TOK / 64, H_HEADS, B_BATCH), ablk(128);
  attn_kernel<<<agrid, ablk, 0, stream>>>(Qb, Kt, Vb, pe, Ob);

  proj_kernel<__bf16, false, false><<<pgrid, pblk, 0, stream>>>(Ob, Wu, bu, (void*)out);
}
